// AttnBlock_36455682409131
// MI455X (gfx1250) — compile-verified, matrix operand order checked by reading, not run on hardware
//
#include <hip/hip_runtime.h>
#include <hip/hip_bf16.h>

typedef __attribute__((ext_vector_type(16))) _Float16 v16h;
typedef __attribute__((ext_vector_type(8)))  _Float16 v8h;
typedef __attribute__((ext_vector_type(4)))  _Float16 v4h;
typedef __attribute__((ext_vector_type(8)))  float    v8f;

#define Bb      4
#define Hs      48
#define Ws      48
#define Nn      2304        // Hs*Ws
#define Cc      256
#define NHEADS  4
#define DH      64
#define MROWS   (Bb*Nn)     // 9216

__device__ __forceinline__ v8f v8f_zero() {
  v8f z = {0.f,0.f,0.f,0.f,0.f,0.f,0.f,0.f};
  return z;
}

// Build a 16-half A/B WMMA fragment: elements 0..7 at p0[0..7] (K=kb..kb+7),
// elements 8..15 at p0[16..23] (K=kb+16..kb+23). 16B-aligned chunks ->
// ds_load_b128 (LDS) or global_load_b128 (global).
__device__ __forceinline__ v16h make_frag(const _Float16* p0) {
  v8h lo = *(const v8h*)(p0);
  v8h hi = *(const v8h*)(p0 + 16);
  v16h r;
#pragma unroll
  for (int i = 0; i < 8; ++i) { r[i] = lo[i]; r[i + 8] = hi[i]; }
  return r;
}

// CDNA5 async global->LDS copy of 16 bytes per lane (ASYNCcnt-tracked).
__device__ __forceinline__ void async_copy_b128(unsigned lds_off, const void* gaddr) {
  asm volatile("global_load_async_to_lds_b128 %0, %1, off"
               :: "v"(lds_off), "v"(gaddr) : "memory");
}
__device__ __forceinline__ void wait_async() {
  asm volatile("s_wait_asynccnt 0x0" ::: "memory");
}
__device__ __forceinline__ unsigned lds_addr32(const void* p) {
  // generic LDS pointer -> wave-relative LDS byte offset (low 32 bits of flat addr)
  return (unsigned)(unsigned long long)p;
}

// ---------------------------------------------------------------------------
// Generic GEMM: C[M,Nout] = A[M,K] @ W[Nout,K]^T  (+ epilogue)
// mode 0: C16 = (f16)acc      (C reinterpreted as _Float16*)
// mode 1: C = acc + res
// mode 2: C = gelu(acc + bias)
// mode 3: C = acc + bias + res
// Block: 128x128 tile, 8 waves (2M x 4N), each wave 64x32 = 4x2 WMMA tiles.
// ---------------------------------------------------------------------------
#define BM 128
#define BN 128
#define BK 32

__global__ __launch_bounds__(256)
void gemm_wmma_kernel(const float* __restrict__ A, const float* __restrict__ Wt,
                      float* __restrict__ C, const float* __restrict__ res,
                      const float* __restrict__ bias,
                      int M, int Nout, int K, int mode) {
  __shared__ __align__(16) _Float16 As[BM * BK];   // [m][k]
  __shared__ __align__(16) _Float16 Bs[BN * BK];   // [n][k]

  const int tid  = threadIdx.x;
  const int lane = tid & 31;
  const int wave = tid >> 5;
  const int hl   = lane >> 4;     // half-lane select
  const int ln   = lane & 15;

  const int m0 = blockIdx.x * BM;
  const int n0 = blockIdx.y * BN;
  const int wm = (wave >> 2) * 64;   // wave row offset in tile
  const int wn = (wave & 3) * 32;    // wave col offset in tile

  v8f acc[4][2];
#pragma unroll
  for (int tm = 0; tm < 4; ++tm)
#pragma unroll
    for (int tn = 0; tn < 2; ++tn) acc[tm][tn] = v8f_zero();

  for (int k0 = 0; k0 < K; k0 += BK) {
    // stage A and B tiles to LDS as f16 (float4 loads, b64 LDS stores)
#pragma unroll
    for (int g = 0; g < 4; ++g) {
      const int i  = tid + g * 256;          // 0..1023
      const int r  = i >> 3;
      const int c4 = (i & 7) * 4;
      float4 fa = *(const float4*)&A [(size_t)(m0 + r) * K + (k0 + c4)];
      float4 fb = *(const float4*)&Wt[(size_t)(n0 + r) * K + (k0 + c4)];
      v4h ha = {(_Float16)fa.x, (_Float16)fa.y, (_Float16)fa.z, (_Float16)fa.w};
      v4h hb = {(_Float16)fb.x, (_Float16)fb.y, (_Float16)fb.z, (_Float16)fb.w};
      *(v4h*)&As[r * BK + c4] = ha;
      *(v4h*)&Bs[r * BK + c4] = hb;
    }
    __syncthreads();

    const int kb = hl * 8;
    v16h af[4], bf[2];
#pragma unroll
    for (int tm = 0; tm < 4; ++tm)
      af[tm] = make_frag(&As[(wm + tm * 16 + ln) * BK + kb]);
#pragma unroll
    for (int tn = 0; tn < 2; ++tn)
      bf[tn] = make_frag(&Bs[(wn + tn * 16 + ln) * BK + kb]);

#pragma unroll
    for (int tm = 0; tm < 4; ++tm)
#pragma unroll
      for (int tn = 0; tn < 2; ++tn)
        acc[tm][tn] = __builtin_amdgcn_wmma_f32_16x16x32_f16(
            false, af[tm], false, bf[tn], (short)0, acc[tm][tn], false, false);
    __syncthreads();
  }

  // epilogue: C-layout -> row = i + 8*hl within 16x16 tile, col = ln
  _Float16* C16 = (_Float16*)C;
#pragma unroll
  for (int tm = 0; tm < 4; ++tm) {
#pragma unroll
    for (int tn = 0; tn < 2; ++tn) {
      const int col = n0 + wn + tn * 16 + ln;
#pragma unroll
      for (int i = 0; i < 8; ++i) {
        const int row = m0 + wm + tm * 16 + i + 8 * hl;
        float v = acc[tm][tn][i];
        const size_t idx = (size_t)row * Nout + col;
        if (mode == 0) {
          C16[idx] = (_Float16)v;
          continue;
        }
        if (mode == 1) {
          v += res[idx];
        } else if (mode == 2) {
          v += bias[col];
          v = 0.5f * v * (1.0f + erff(v * 0.70710678118f));
        } else if (mode == 3) {
          v += bias[col] + res[idx];
        }
        C[idx] = v;
      }
    }
  }
}

// ---------------------------------------------------------------------------
// RoPE-2D in place on q/k slices of the f16 qkv buffer [MROWS, 768].
// One block per (b,n); thread -> one even/odd pair. Math in f32.
// ---------------------------------------------------------------------------
__global__ __launch_bounds__(256)
void rope2d_kernel(_Float16* __restrict__ qkv) {
  const int rowid = blockIdx.x;            // 0..MROWS-1
  const int n = rowid % Nn;
  const float ycoord = (float)(n / Ws);
  const float xcoord = (float)(n % Ws);

  const int t   = threadIdx.x;
  const int s   = t >> 7;          // 0 = q, 1 = k
  const int rem = t & 127;
  const int h   = rem >> 5;
  const int p   = rem & 31;
  const int d0  = 2 * p;           // even dim in [0,64)

  const float coord = (d0 < 32) ? ycoord : xcoord;
  const int   f0    = d0 & 31;
  const float a0 = coord * __powf(10000.0f, -(float)f0 / 32.0f);
  const float a1 = coord * __powf(10000.0f, -(float)(f0 + 1) / 32.0f);

  _Float16* ptr = qkv + (size_t)rowid * (3 * Cc) + s * Cc + h * DH + d0;
  const float t0 = (float)ptr[0];
  const float t1 = (float)ptr[1];
  ptr[0] = (_Float16)(t0 * __cosf(a0) - t1 * __sinf(a0));
  ptr[1] = (_Float16)(t1 * __cosf(a1) + t0 * __sinf(a1));
}

// ---------------------------------------------------------------------------
// Flash attention: grid (N/128, B*NHEADS), block 256 (8 waves x 16 q-rows).
// K tile staged with CDNA5 async global->LDS b128 copies; V staged transposed.
// ---------------------------------------------------------------------------
__global__ __launch_bounds__(256)
void attn_kernel(const _Float16* __restrict__ qkv, float* __restrict__ attn_out) {
  __shared__ __align__(16) _Float16 Ks [128 * DH];     // [key][d]
  __shared__ __align__(16) _Float16 VsT[DH * 128];     // [d][key]
  __shared__ __align__(16) _Float16 Ps [8 * 16 * 128]; // per-wave P tiles

  const int tid  = threadIdx.x;
  const int lane = tid & 31;
  const int wave = tid >> 5;
  const int hl   = lane >> 4;
  const int ln   = lane & 15;

  const int bh = blockIdx.y;
  const int b  = bh >> 2;
  const int h  = bh & 3;
  const int q0 = blockIdx.x * 128 + wave * 16;   // wave's first query row

  const _Float16* Qg = qkv + (size_t)b * Nn * (3 * Cc) + h * DH;
  const _Float16* Kg = Qg + Cc;
  const _Float16* Vg = Qg + 2 * Cc;

  const unsigned ks_base = lds_addr32(&Ks[0]);

  // Q A-fragments (two k-steps of 32): raw 16B global loads, no scaling
  v16h qf[2];
#pragma unroll
  for (int ks = 0; ks < 2; ++ks)
    qf[ks] = make_frag(Qg + (size_t)(q0 + ln) * (3 * Cc) + hl * 8 + ks * 32);

  v8f Oacc[4];
#pragma unroll
  for (int vt = 0; vt < 4; ++vt) Oacc[vt] = v8f_zero();
  float mrow[8], lrow[8];
#pragma unroll
  for (int i = 0; i < 8; ++i) { mrow[i] = -3.0e38f; lrow[i] = 0.f; }

  _Float16* Pw = &Ps[wave * 16 * 128];

  for (int kt = 0; kt < Nn / 128; ++kt) {
    const int kv0 = kt * 128;

    // K tile: pure byte copy -> async to LDS (8 halves = 16B per chunk)
#pragma unroll
    for (int g = 0; g < 4; ++g) {
      const int c = tid + g * 256;          // 0..1023 chunks
      const int r = c >> 3, pt = c & 7;
      async_copy_b128(ks_base + (unsigned)c * 16,
                      Kg + (size_t)(kv0 + r) * (3 * Cc) + pt * 8);
    }
    // V tile: transpose while staging (v8h global loads, b16 LDS scatter)
#pragma unroll
    for (int g = 0; g < 4; ++g) {
      const int c = tid + g * 256;
      const int r = c >> 3, pt = c & 7;
      v8h vv = *(const v8h*)(Vg + (size_t)(kv0 + r) * (3 * Cc) + pt * 8);
#pragma unroll
      for (int j = 0; j < 8; ++j)
        VsT[(pt * 8 + j) * 128 + r] = vv[j];
    }
    wait_async();
    __syncthreads();

    // S = Q @ K^T : 8 n-tiles of 16 keys
    v8f S[8];
#pragma unroll
    for (int nt = 0; nt < 8; ++nt) {
      v8f s = v8f_zero();
#pragma unroll
      for (int ks = 0; ks < 2; ++ks) {
        v16h bfr = make_frag(&Ks[(nt * 16 + ln) * DH + hl * 8 + ks * 32]);
        s = __builtin_amdgcn_wmma_f32_16x16x32_f16(
            false, qf[ks], false, bfr, (short)0, s, false, false);
      }
      S[nt] = s;
    }
    // apply 1/sqrt(dh)
#pragma unroll
    for (int nt = 0; nt < 8; ++nt)
#pragma unroll
      for (int i = 0; i < 8; ++i) S[nt][i] *= 0.125f;

    // online softmax per row (row i lives in VGPR i across 16-lane half)
#pragma unroll
    for (int i = 0; i < 8; ++i) {
      float mx = S[0][i];
#pragma unroll
      for (int nt = 1; nt < 8; ++nt) mx = fmaxf(mx, S[nt][i]);
#pragma unroll
      for (int off = 1; off < 16; off <<= 1) mx = fmaxf(mx, __shfl_xor(mx, off, 32));
      const float mnew = fmaxf(mrow[i], mx);
      const float corr = __expf(mrow[i] - mnew);
      mrow[i] = mnew;
      lrow[i] *= corr;
#pragma unroll
      for (int vt = 0; vt < 4; ++vt) Oacc[vt][i] *= corr;
      float rs = 0.f;
#pragma unroll
      for (int nt = 0; nt < 8; ++nt) {
        const float e = __expf(S[nt][i] - mnew);
        S[nt][i] = e;
        rs += e;
      }
#pragma unroll
      for (int off = 1; off < 16; off <<= 1) rs += __shfl_xor(rs, off, 32);
      lrow[i] += rs;
    }

    // C-layout -> A-layout reshape of P through wave-private LDS
#pragma unroll
    for (int nt = 0; nt < 8; ++nt)
#pragma unroll
      for (int i = 0; i < 8; ++i)
        Pw[(i + 8 * hl) * 128 + nt * 16 + ln] = (_Float16)S[nt][i];

    // O += P @ V
#pragma unroll
    for (int ks = 0; ks < 4; ++ks) {
      v16h af = make_frag(&Pw[ln * 128 + hl * 8 + ks * 32]);
#pragma unroll
      for (int vt = 0; vt < 4; ++vt) {
        v16h bfr = make_frag(&VsT[(vt * 16 + ln) * 128 + hl * 8 + ks * 32]);
        Oacc[vt] = __builtin_amdgcn_wmma_f32_16x16x32_f16(
            false, af, false, bfr, (short)0, Oacc[vt], false, false);
      }
    }
    __syncthreads();
  }

  // write O / l  to (B,N,C) layout (f32)
  float* outp = attn_out + (size_t)b * Nn * Cc + h * DH;
#pragma unroll
  for (int vt = 0; vt < 4; ++vt)
#pragma unroll
    for (int i = 0; i < 8; ++i) {
      const int row = q0 + i + 8 * hl;
      outp[(size_t)row * Cc + vt * 16 + ln] = Oacc[vt][i] / lrow[i];
    }
}

// ---------------------------------------------------------------------------
// LayerNorm over C=256, one block per row.
// ---------------------------------------------------------------------------
__global__ __launch_bounds__(256)
void layernorm_kernel(const float* __restrict__ in, const float* __restrict__ g,
                      const float* __restrict__ bta, float* __restrict__ out) {
  __shared__ float r1[8], r2[8];
  const int row = blockIdx.x;
  const int t   = threadIdx.x;
  const float v = in[(size_t)row * Cc + t];

  float s1 = v, s2 = v * v;
#pragma unroll
  for (int off = 16; off >= 1; off >>= 1) {
    s1 += __shfl_xor(s1, off, 32);
    s2 += __shfl_xor(s2, off, 32);
  }
  if ((t & 31) == 0) { r1[t >> 5] = s1; r2[t >> 5] = s2; }
  __syncthreads();
  float tot1 = 0.f, tot2 = 0.f;
#pragma unroll
  for (int i = 0; i < 8; ++i) { tot1 += r1[i]; tot2 += r2[i]; }
  const float mu  = tot1 * (1.0f / Cc);
  const float var = tot2 * (1.0f / Cc) - mu * mu;
  const float inv = rsqrtf(var + 1e-5f);
  out[(size_t)row * Cc + t] = (v - mu) * inv * g[t] + bta[t];
}

// ---------------------------------------------------------------------------
extern "C" void kernel_launch(void* const* d_in, const int* in_sizes, int n_in,
                              void* d_out, int out_size, void* d_ws, size_t ws_size,
                              hipStream_t stream) {
  const float* x     = (const float*)d_in[0];
  const float* Wqkv  = (const float*)d_in[1];
  const float* Wproj = (const float*)d_in[2];
  const float* g1    = (const float*)d_in[3];
  const float* b1    = (const float*)d_in[4];
  const float* g2    = (const float*)d_in[5];
  const float* b2    = (const float*)d_in[6];
  const float* W1    = (const float*)d_in[7];
  const float* bf1   = (const float*)d_in[8];
  const float* W2    = (const float*)d_in[9];
  const float* bf2   = (const float*)d_in[10];
  (void)in_sizes; (void)n_in; (void)out_size; (void)ws_size;

  _Float16* qkv16 = (_Float16*)d_ws;                  // [MROWS, 768] f16
  float* wsf  = (float*)(qkv16 + (size_t)MROWS * 3 * Cc);
  float* attn = wsf;                                  // [MROWS, 256]
  float* ybuf = attn + (size_t)MROWS * Cc;            // [MROWS, 256]
  float* x1   = ybuf + (size_t)MROWS * Cc;            // [MROWS, 256]
  float* hbuf = x1   + (size_t)MROWS * Cc;            // [MROWS, 1024]
  float* zbuf = hbuf + (size_t)MROWS * 4 * Cc;        // [MROWS, 256]
  float* outp = (float*)d_out;

  const dim3 blk(256);
  const int mt = MROWS / BM;   // 72

  // 1) qkv(f16) = x @ Wqkv^T
  gemm_wmma_kernel<<<dim3(mt, (3 * Cc) / BN), blk, 0, stream>>>(
      x, Wqkv, (float*)qkv16, nullptr, nullptr, MROWS, 3 * Cc, Cc, 0);
  // 2) RoPE-2D in place on q,k
  rope2d_kernel<<<dim3(MROWS), blk, 0, stream>>>(qkv16);
  // 3) attention
  attn_kernel<<<dim3(Nn / 128, Bb * NHEADS), blk, 0, stream>>>(qkv16, attn);
  // 4) y = x + attn @ Wproj^T
  gemm_wmma_kernel<<<dim3(mt, Cc / BN), blk, 0, stream>>>(
      attn, Wproj, ybuf, x, nullptr, MROWS, Cc, Cc, 1);
  // 5) x1 = LN(y)
  layernorm_kernel<<<dim3(MROWS), blk, 0, stream>>>(ybuf, g1, b1, x1);
  // 6) h = gelu(x1 @ W1^T + bf1)
  gemm_wmma_kernel<<<dim3(mt, (4 * Cc) / BN), blk, 0, stream>>>(
      x1, W1, hbuf, nullptr, bf1, MROWS, 4 * Cc, Cc, 2);
  // 7) z = x1 + h @ W2^T + bf2
  gemm_wmma_kernel<<<dim3(mt, Cc / BN), blk, 0, stream>>>(
      hbuf, W2, zbuf, x1, bf2, MROWS, Cc, 4 * Cc, 3);
  // 8) out = LN(z)
  layernorm_kernel<<<dim3(MROWS), blk, 0, stream>>>(zbuf, g2, b2, outp);
}
